// GCN_45586782880364
// MI455X (gfx1250) — matrix operand order repaired and checked by reading, not compiled, not run on hardware
//
#include <hip/hip_runtime.h>
#include <hip/hip_bf16.h>

typedef __attribute__((ext_vector_type(16))) _Float16 v16h;
typedef __attribute__((ext_vector_type(8)))  float    v8f;

#define DIM 128

// ---------------------------------------------------------------------------
// Degree / normalization kernels
// ---------------------------------------------------------------------------
__global__ void gcn_deg_init(float* deg, int n) {
    int i = blockIdx.x * blockDim.x + threadIdx.x;
    if (i < n) deg[i] = 1.0f;  // self-loop contributes 1 to every node's degree
}

__global__ void gcn_deg_accum(const int* __restrict__ dst, float* deg, int e) {
    int i = blockIdx.x * blockDim.x + threadIdx.x;
    if (i < e) atomicAdd(&deg[dst[i]], 1.0f);
}

__global__ void gcn_deg_to_dinv(float* deg, int n) {
    int i = blockIdx.x * blockDim.x + threadIdx.x;
    if (i < n) {
        float d = deg[i];               // >= 1 always (self loop)
        deg[i] = rsqrtf(d);
    }
}

__global__ void gcn_edge_norm(const int* __restrict__ src, const int* __restrict__ dst,
                              const float* __restrict__ dinv, float* __restrict__ norm, int e) {
    int i = blockIdx.x * blockDim.x + threadIdx.x;
    if (i < e) norm[i] = dinv[src[i]] * dinv[dst[i]];
}

// ---------------------------------------------------------------------------
// GEMM: C[M,128] = A[M,128] @ W[128,128]  via v_wmma_f32_16x16x32_f16
// One wave owns a 16-row strip; 8 waves (128 rows) per block.
// W is staged into LDS pre-swizzled into a contiguous-K B-fragment layout:
//   Ws[ntile][kchunk][lane][0..15] = 16 contiguous K values for this lane's
//   column, so the inner-loop B load is a single aligned 32B LDS read.
// ---------------------------------------------------------------------------
__global__ __launch_bounds__(256) void gcn_gemm_wmma(const float* __restrict__ A,
                                                     const float* __restrict__ W,
                                                     float* __restrict__ C, int M) {
    __shared__ __align__(32) _Float16 Ws[8 * 4 * 32 * 16];  // 32 KB

    // Stage W (fp32 -> f16) into B-fragment layout.
    for (int idx = threadIdx.x; idx < DIM * DIM; idx += 256) {
        int k = idx >> 7;          // row of W (K dimension)
        int n = idx & 127;         // col of W (N dimension)
        int ntile  = n >> 4;       // which 16-col output tile
        int kchunk = k >> 5;       // which K=32 chunk
        int lane   = (n & 15) | (((k >> 4) & 1) << 4);  // lane halves split K 0-15 / 16-31
        int off    = k & 15;
        Ws[(((ntile * 4 + kchunk) * 32 + lane) * 16) + off] = (_Float16)W[idx];
    }
    __syncthreads();

    const int wave = threadIdx.x >> 5;
    const int lane = threadIdx.x & 31;
    const int strip = blockIdx.x * 8 + wave;     // 16-row strip
    if (strip * 16 >= M) return;                 // wave-uniform exit (EXEC all-1 for WMMA)

    const int mrow = strip * 16 + (lane & 15);   // A row for this lane
    const int half = lane >> 4;                  // K-half select within chunk

    // Load the four 16x32 f16 A fragments for this strip (contiguous-K order, not ISA table order):
    // lane holds row M = lane%16; VGPRs 0-7 = K run [half*16, half*16+16)
    // (within each 32-K chunk), matching the B staging.
    v16h a[4];
    const float* arow = A + (size_t)mrow * DIM;
    #pragma unroll
    for (int c = 0; c < 4; ++c) {
        const int base = c * 32 + half * 16;
        #pragma unroll
        for (int j = 0; j < 8; ++j) {
            a[c][j]     = (_Float16)arow[base + j];
            a[c][8 + j] = (_Float16)arow[base + 8 + j];
        }
    }

    // 8 output tiles along N; 4 chained WMMAs (K=128) per tile.
    #pragma unroll
    for (int t = 0; t < 8; ++t) {
        v8f acc = {};
        #pragma unroll
        for (int c = 0; c < 4; ++c) {
            v16h b = *(const v16h*)&Ws[((t * 4 + c) * 32 + lane) * 16];
            acc = __builtin_amdgcn_wmma_f32_16x16x32_f16(
                false, a[c], false, b, (short)0, acc, false, false);
        }
        // C/D layout: lane -> col = lane%16, rows = strip*16 + half*8 + r
        const int col   = t * 16 + (lane & 15);
        const int rbase = strip * 16 + half * 8;
        #pragma unroll
        for (int r = 0; r < 8; ++r) {
            C[(size_t)(rbase + r) * DIM + col] = acc[r];
        }
    }
}

// ---------------------------------------------------------------------------
// Aggregation: acc[i] = xl[i]*dinv[i]^2 (self loop), then scatter-add edges.
// One 32-lane group per node/edge, float4 per lane (128 cols).
// ---------------------------------------------------------------------------
__global__ __launch_bounds__(256) void gcn_acc_init(const float* __restrict__ xl,
                                                    const float* __restrict__ dinv,
                                                    float* __restrict__ acc, int n) {
    int t = blockIdx.x * blockDim.x + threadIdx.x;
    int i = t >> 5;
    if (i >= n) return;
    int lane = t & 31;
    float w = dinv[i];
    w = w * w;
    const float4 v = *(const float4*)(xl + (size_t)i * DIM + lane * 4);
    float4 r = make_float4(v.x * w, v.y * w, v.z * w, v.w * w);
    *(float4*)(acc + (size_t)i * DIM + lane * 4) = r;
}

__global__ __launch_bounds__(256) void gcn_edge_agg(const float* __restrict__ xl,
                                                    const int* __restrict__ src,
                                                    const int* __restrict__ dst,
                                                    const float* __restrict__ norm,
                                                    float* __restrict__ acc, int e) {
    int t = blockIdx.x * blockDim.x + threadIdx.x;
    int ei = t >> 5;
    if (ei >= e) return;
    int lane = t & 31;
    int s = src[ei];
    int d = dst[ei];
    float nm = norm[ei];
    const float4 v = *(const float4*)(xl + (size_t)s * DIM + lane * 4);
    float* o = acc + (size_t)d * DIM + lane * 4;
    atomicAdd(o + 0, v.x * nm);   // f32 atomics resolve in L2 (51MB set fits 192MB L2)
    atomicAdd(o + 1, v.y * nm);
    atomicAdd(o + 2, v.z * nm);
    atomicAdd(o + 3, v.w * nm);
}

__global__ __launch_bounds__(256) void gcn_bias_relu(const float* __restrict__ acc,
                                                     const float* __restrict__ bias,
                                                     float* __restrict__ out, int n) {
    int t = blockIdx.x * blockDim.x + threadIdx.x;
    int i = t >> 5;
    if (i >= n) return;
    int lane = t & 31;
    const float4 b = *(const float4*)(bias + lane * 4);
    const float4 v = *(const float4*)(acc + (size_t)i * DIM + lane * 4);
    float4 r = make_float4(fmaxf(v.x + b.x, 0.0f), fmaxf(v.y + b.y, 0.0f),
                           fmaxf(v.z + b.z, 0.0f), fmaxf(v.w + b.w, 0.0f));
    *(float4*)(out + (size_t)i * DIM + lane * 4) = r;
}

// ---------------------------------------------------------------------------
// Driver
// ---------------------------------------------------------------------------
extern "C" void kernel_launch(void* const* d_in, const int* in_sizes, int n_in,
                              void* d_out, int out_size, void* d_ws, size_t ws_size,
                              hipStream_t stream) {
    const float* x   = (const float*)d_in[0];
    const int*   ei  = (const int*)d_in[1];
    const float* W1  = (const float*)d_in[2];
    const float* b1  = (const float*)d_in[3];
    const float* W2  = (const float*)d_in[4];
    const float* b2  = (const float*)d_in[5];
    float*       out = (float*)d_out;

    const int N = in_sizes[0] / DIM;       // 100000
    const int E = in_sizes[1] / 2;         // 1600000
    const int* src = ei;                   // edge_index[0]
    const int* dst = ei + E;               // edge_index[1]

    // Workspace layout (floats): dinv | norm | B1 | B2
    float* dinv = (float*)d_ws;
    float* norm = dinv + ((N + 1023) & ~1023);
    float* B1   = norm + ((E + 1023) & ~1023);     // 51.2 MB buffer
    float* B2   = B1 + (size_t)N * DIM;            // 51.2 MB buffer

    const int TPB = 256;
    const int gN   = (N + TPB - 1) / TPB;
    const int gE   = (E + TPB - 1) / TPB;
    const int gN32 = (N * 32 + TPB - 1) / TPB;     // one 32-lane group per node
    const int gE32 = (E * 32 + TPB - 1) / TPB;     // one 32-lane group per edge
    const int strips = (N + 15) / 16;
    const int gGemm  = (strips + 7) / 8;           // 8 waves (strips) per block

    // Degree -> dinv -> per-edge norm (reused by both layers)
    gcn_deg_init<<<gN, TPB, 0, stream>>>(dinv, N);
    gcn_deg_accum<<<gE, TPB, 0, stream>>>(dst, dinv, E);
    gcn_deg_to_dinv<<<gN, TPB, 0, stream>>>(dinv, N);
    gcn_edge_norm<<<gE, TPB, 0, stream>>>(src, dst, dinv, norm, E);

    // ---- Layer 1: h = relu(Agg(x @ W1) + b1) ----
    gcn_gemm_wmma<<<gGemm, TPB, 0, stream>>>(x, W1, B2, N);          // B2 = x @ W1
    gcn_acc_init<<<gN32, TPB, 0, stream>>>(B2, dinv, B1, N);         // B1 = self-loop term
    gcn_edge_agg<<<gE32, TPB, 0, stream>>>(B2, src, dst, norm, B1, E);
    gcn_bias_relu<<<gN32, TPB, 0, stream>>>(B1, b1, B1, N);          // B1 = h (in place)

    // ---- Layer 2: out = relu(Agg(h @ W2) + b2) ----
    gcn_gemm_wmma<<<gGemm, TPB, 0, stream>>>(B1, W2, B2, N);         // B2 = h @ W2
    gcn_acc_init<<<gN32, TPB, 0, stream>>>(B2, dinv, B1, N);         // reuse B1 as acc
    gcn_edge_agg<<<gE32, TPB, 0, stream>>>(B2, src, dst, norm, B1, E);
    gcn_bias_relu<<<gN32, TPB, 0, stream>>>(B1, b2, out, N);
}
